// ColPredictor_37125697307275
// MI455X (gfx1250) — matrix operand order repaired and checked by reading, not compiled, not run on hardware
//
#include <hip/hip_runtime.h>
#include <math.h>

// ---------------- problem constants ----------------
#define BB   128
#define QQ   64
#define HSQ  32
#define CC   128
#define DD   300
#define HH   512
#define G4   2048   // 4*H
#define NLAB 4

// ---------------- WMMA GEMM ----------------
// C[z] = A[z] @ op(B[z]) (+bias +bias2) (+= existing C if accum)
//   A: row-major [M,K], lda
//   B: if b_is_nk: row-major [N,K] (computes A@B^T);  else row-major [K,N] (computes A@B)
//   C: row-major [M,N], ldc
// Block tile 64x64, K-chunk 32, 128 threads = 4 wave32 waves (2 in M x 2 in N),
// each wave owns a 32x32 C sub-tile = four v_wmma_f32_16x16x32_bf16 accumulators.
// Full tiles take a double-buffered pipelined path (1 barrier / chunk, global
// fetch of chunk k+1 overlapped with WMMAs of chunk k).
typedef __attribute__((ext_vector_type(16))) __bf16 v16bf;
typedef __attribute__((ext_vector_type(4)))  __bf16 v4bf;
typedef __attribute__((ext_vector_type(8)))  float  v8f;

#define BM 64
#define BN 64
#define BK 32
#define LDST 40   // padded LDS row stride (bf16 elems); 80B rows keep 8B alignment

__global__ __launch_bounds__(128)
void wmma_gemm(const float* __restrict__ A, const float* __restrict__ Bm,
               const float* __restrict__ bias, const float* __restrict__ bias2,
               float* __restrict__ C,
               int M, int N, int K, int lda, int ldb, int ldc,
               long long sA, long long sB, long long sC,
               int b_is_nk, int accum)
{
    __shared__ alignas(16) __bf16 lA[2][BM][LDST];
    __shared__ alignas(16) __bf16 lB[2][BN][LDST];   // indexed [n][k]

    const int z = blockIdx.z;
    A  += (long long)z * sA;
    Bm += (long long)z * sB;
    C  += (long long)z * sC;

    const int m0   = blockIdx.y * BM;
    const int n0   = blockIdx.x * BN;
    const int tid  = threadIdx.x;
    const int lane = tid & 31;
    const int w    = tid >> 5;     // 0..3
    const int wm   = w >> 1;       // 0..1  (32-row band)
    const int wn   = w & 1;        // 0..1  (32-col band)
    const int lm   = lane & 15;
    const int half = lane >> 4;

    const bool fullA = (m0 + BM <= M);
    const bool fullB = (n0 + BN <= N);

    v8f acc00 = {}, acc01 = {}, acc10 = {}, acc11 = {};

    // gather fragments from LDS buffer `buf` and issue 4 WMMAs
    auto computeChunk = [&](int buf) {
        v16bf a0, a1, b0, b1;
#pragma unroll
        for (int v = 0; v < 8; ++v) {
            int ka = (v < 4) ? (half * 8 + v * 2) : (16 + half * 8 + (v - 4) * 2);
            a0[2 * v]     = lA[buf][wm * 32 + lm][ka];
            a0[2 * v + 1] = lA[buf][wm * 32 + lm][ka + 1];
            a1[2 * v]     = lA[buf][wm * 32 + 16 + lm][ka];
            a1[2 * v + 1] = lA[buf][wm * 32 + 16 + lm][ka + 1];
            int kb = (v < 4) ? (half * 8 + v * 2) : (16 + half * 8 + (v - 4) * 2);
            b0[2 * v]     = lB[buf][wn * 32 + lm][kb];
            b0[2 * v + 1] = lB[buf][wn * 32 + lm][kb + 1];
            b1[2 * v]     = lB[buf][wn * 32 + 16 + lm][kb];
            b1[2 * v + 1] = lB[buf][wn * 32 + 16 + lm][kb + 1];
        }
        acc00 = __builtin_amdgcn_wmma_f32_16x16x32_bf16(false, a0, false, b0, (short)0, acc00, false, false);
        acc01 = __builtin_amdgcn_wmma_f32_16x16x32_bf16(false, a0, false, b1, (short)0, acc01, false, false);
        acc10 = __builtin_amdgcn_wmma_f32_16x16x32_bf16(false, a1, false, b0, (short)0, acc10, false, false);
        acc11 = __builtin_amdgcn_wmma_f32_16x16x32_bf16(false, a1, false, b1, (short)0, acc11, false, false);
    };

    if (fullA && fullB && (K & (BK - 1)) == 0) {
        // ---------- pipelined double-buffered fast path (all tiles full) ----------
        float4 ra[4], rb[4];
        auto ldA = [&](int k0) {
#pragma unroll
            for (int i = 0; i < 4; ++i) {
                int e = tid + i * 128;
                int r = e >> 3, q = e & 7;
                ra[i] = *(const float4*)(A + (long long)(m0 + r) * lda + k0 + q * 4);
            }
        };
        auto stA = [&](int buf) {
#pragma unroll
            for (int i = 0; i < 4; ++i) {
                int e = tid + i * 128;
                int r = e >> 3, q = e & 7;
                v4bf o;
                o[0] = (__bf16)ra[i].x; o[1] = (__bf16)ra[i].y;
                o[2] = (__bf16)ra[i].z; o[3] = (__bf16)ra[i].w;
                *(v4bf*)&lA[buf][r][q * 4] = o;
            }
        };
        auto ldB = [&](int k0) {
#pragma unroll
            for (int i = 0; i < 4; ++i) {
                int e = tid + i * 128;
                if (b_is_nk) {
                    int r = e >> 3, q = e & 7;
                    rb[i] = *(const float4*)(Bm + (long long)(n0 + r) * ldb + k0 + q * 4);
                } else {
                    int kk = e >> 4, nq = (e & 15) * 4;
                    rb[i] = *(const float4*)(Bm + (long long)(k0 + kk) * ldb + n0 + nq);
                }
            }
        };
        auto stB = [&](int buf) {
#pragma unroll
            for (int i = 0; i < 4; ++i) {
                int e = tid + i * 128;
                if (b_is_nk) {
                    int r = e >> 3, q = e & 7;
                    v4bf o;
                    o[0] = (__bf16)rb[i].x; o[1] = (__bf16)rb[i].y;
                    o[2] = (__bf16)rb[i].z; o[3] = (__bf16)rb[i].w;
                    *(v4bf*)&lB[buf][r][q * 4] = o;
                } else {
                    int kk = e >> 4, nq = (e & 15) * 4;
                    lB[buf][nq + 0][kk] = (__bf16)rb[i].x;
                    lB[buf][nq + 1][kk] = (__bf16)rb[i].y;
                    lB[buf][nq + 2][kk] = (__bf16)rb[i].z;
                    lB[buf][nq + 3][kk] = (__bf16)rb[i].w;
                }
            }
        };

        ldA(0); ldB(0);
        stA(0); stB(0);
        __syncthreads();
        const int nk = K / BK;
        for (int kc = 0; kc < nk; ++kc) {
            const int cur  = kc & 1;
            const bool more = (kc + 1 < nk);
            if (more) { ldA((kc + 1) * BK); ldB((kc + 1) * BK); }   // overlap with WMMAs
            computeChunk(cur);
            if (more) { stA(cur ^ 1); stB(cur ^ 1); }
            __syncthreads();
        }
    } else {
        // ---------- generic guarded path (tails / partial tiles) ----------
        for (int k0 = 0; k0 < K; k0 += BK) {
            const bool fullK = (k0 + BK <= K);

            if (fullA && fullK) {
                if (k0 + BK < K)   // global_prefetch_b8 of next chunk
                    __builtin_prefetch(A + (long long)(m0 + (tid >> 1)) * lda + k0 + BK, 0, 0);
#pragma unroll
                for (int i = 0; i < 4; ++i) {
                    int e = tid + i * 128;
                    int r = e >> 3, q = e & 7;
                    const float4 v = *(const float4*)(A + (long long)(m0 + r) * lda + k0 + q * 4);
                    v4bf o;
                    o[0] = (__bf16)v.x; o[1] = (__bf16)v.y; o[2] = (__bf16)v.z; o[3] = (__bf16)v.w;
                    *(v4bf*)&lA[0][r][q * 4] = o;
                }
            } else {
                for (int e = tid; e < BM * BK; e += 128) {
                    int r = e >> 5, c = e & 31;
                    int gm = m0 + r, gk = k0 + c;
                    float v = (gm < M && gk < K) ? A[(long long)gm * lda + gk] : 0.0f;
                    lA[0][r][c] = (__bf16)v;
                }
            }

            if (b_is_nk) {
                if (fullB && fullK) {
#pragma unroll
                    for (int i = 0; i < 4; ++i) {
                        int e = tid + i * 128;
                        int r = e >> 3, q = e & 7;
                        const float4 v = *(const float4*)(Bm + (long long)(n0 + r) * ldb + k0 + q * 4);
                        v4bf o;
                        o[0] = (__bf16)v.x; o[1] = (__bf16)v.y; o[2] = (__bf16)v.z; o[3] = (__bf16)v.w;
                        *(v4bf*)&lB[0][r][q * 4] = o;
                    }
                } else {
                    for (int e = tid; e < BN * BK; e += 128) {
                        int r = e >> 5, c = e & 31;           // r=n, c=k
                        int gn = n0 + r, gk = k0 + c;
                        float v = (gn < N && gk < K) ? Bm[(long long)gn * ldb + gk] : 0.0f;
                        lB[0][r][c] = (__bf16)v;
                    }
                }
            } else {
                if (fullB && fullK) {
#pragma unroll
                    for (int i = 0; i < 4; ++i) {
                        int e = tid + i * 128;
                        int kk = e >> 4, nq = (e & 15) * 4;
                        const float4 v = *(const float4*)(Bm + (long long)(k0 + kk) * ldb + n0 + nq);
                        lB[0][nq + 0][kk] = (__bf16)v.x;
                        lB[0][nq + 1][kk] = (__bf16)v.y;
                        lB[0][nq + 2][kk] = (__bf16)v.z;
                        lB[0][nq + 3][kk] = (__bf16)v.w;
                    }
                } else {
                    for (int e = tid; e < BN * BK; e += 128) {
                        int r = e & 63, c = e >> 6;           // r=n, c=k
                        int gn = n0 + r, gk = k0 + c;
                        float v = (gn < N && gk < K) ? Bm[(long long)gk * ldb + gn] : 0.0f;
                        lB[0][r][c] = (__bf16)v;
                    }
                }
            }
            __syncthreads();
            computeChunk(0);
            __syncthreads();
        }
    }

    // ---- store D (lane%16 = N, VGPR r -> M = r + half*8) ----
    auto storeTile = [&](const v8f& acc, int msub, int nsub) {
#pragma unroll
        for (int r = 0; r < 8; ++r) {
            int gm = m0 + wm * 32 + msub + half * 8 + r;
            int gn = n0 + wn * 32 + nsub + lm;
            if (gm < M && gn < N) {
                float vv = acc[r];
                if (bias)  vv += bias[gn];
                if (bias2) vv += bias2[gn];
                if (accum) vv += C[(long long)gm * ldc + gn];
                C[(long long)gm * ldc + gn] = vv;
            }
        }
    };
    storeTile(acc00, 0,  0);
    storeTile(acc01, 0,  16);
    storeTile(acc10, 16, 0);
    storeTile(acc11, 16, 16);
}

// ---------------- elementwise / small kernels ----------------
__global__ void zero_f32(float* p, int n) {
    int i = blockIdx.x * blockDim.x + threadIdx.x;
    if (i < n) p[i] = 0.0f;
}

__global__ void add2_f32(float* out, const float* a, const float* b, int n) {
    int i = blockIdx.x * blockDim.x + threadIdx.x;
    if (i < n) out[i] = a[i] + b[i];
}

// Fused LSTM cell (torch gate order i,f,g,o): gates = xg_t + h@Whh^T (precomputed)
__global__ void lstm_cell(const float* __restrict__ xg, long long xg_row_stride,
                          const float* __restrict__ ghh,
                          float* __restrict__ c, float* __restrict__ h,
                          float* __restrict__ enc_t, long long enc_row_stride,
                          int Bsz, int Hn)
{
    int i = blockIdx.x * blockDim.x + threadIdx.x;
    if (i >= Bsz * Hn) return;
    int b = i / Hn, j = i % Hn;
    const float* xr = xg + (long long)b * xg_row_stride;
    const float* gr = ghh + (long long)b * 4 * Hn;
    float gi = xr[j]          + gr[j];
    float gf = xr[Hn + j]     + gr[Hn + j];
    float gg = xr[2 * Hn + j] + gr[2 * Hn + j];
    float go = xr[3 * Hn + j] + gr[3 * Hn + j];
    float si = 1.0f / (1.0f + __expf(-gi));
    float sf = 1.0f / (1.0f + __expf(-gf));
    float so = 1.0f / (1.0f + __expf(-go));
    float cv = sf * c[i] + si * tanhf(gg);
    float hv = so * tanhf(cv);
    c[i] = cv;
    h[i] = hv;
    enc_t[(long long)b * enc_row_stride + j] = hv;
}

__global__ void gather_rows(float* __restrict__ dst, const float* __restrict__ src,
                            const int* __restrict__ recover, long long rowElems,
                            long long n)
{
    long long i = (long long)blockIdx.x * blockDim.x + threadIdx.x;
    if (i >= n) return;
    int b = (int)(i / rowElems);
    long long r = i % rowElems;
    dst[i] = src[(long long)recover[b] * rowElems + r];
}

// masked softmax over last dim, in place; row = (b, c)
__global__ void softmax_mask(float* __restrict__ att, int Cdim, int L,
                             const int* __restrict__ col_len, const int* __restrict__ col_rec,
                             const int* __restrict__ len2, const int* __restrict__ rec2)
{
    int row = blockIdx.x;
    int b = row / Cdim, c = row % Cdim;
    float* p = att + (long long)row * L;
    bool cvalid = c < col_len[col_rec[b]];
    int  l2 = len2[rec2[b]];
    __shared__ float red[64];
    int tid = threadIdx.x;

    float mx = -1e30f;
    for (int l = tid; l < L; l += 64) {
        bool valid = cvalid && (l < l2);
        float v = valid ? p[l] : -100.0f;
        p[l] = v;
        mx = fmaxf(mx, v);
    }
    red[tid] = mx; __syncthreads();
    for (int s = 32; s > 0; s >>= 1) { if (tid < s) red[tid] = fmaxf(red[tid], red[tid + s]); __syncthreads(); }
    mx = red[0]; __syncthreads();

    float sum = 0.0f;
    for (int l = tid; l < L; l += 64) { float e = __expf(p[l] - mx); p[l] = e; sum += e; }
    red[tid] = sum; __syncthreads();
    for (int s = 32; s > 0; s >>= 1) { if (tid < s) red[tid] += red[tid + s]; __syncthreads(); }
    float inv = 1.0f / red[0];
    for (int l = tid; l < L; l += 64) p[l] *= inv;
}

// s[b,l] = sum_c prob[b,c,l]
__global__ void sum_over_c(const float* __restrict__ prob, float* __restrict__ s,
                           int Cdim, int L, int n)
{
    int i = blockIdx.x * blockDim.x + threadIdx.x;
    if (i >= n) return;
    int b = i / L, q = i % L;
    float acc = 0.0f;
    for (int c = 0; c < Cdim; ++c)
        acc += prob[((long long)b * Cdim + c) * L + q];
    s[i] = acc;
}

// out[b,h] = sum_t s[b,t] * enc[b,t,h]
__global__ void weighted_sum(const float* __restrict__ s, const float* __restrict__ enc,
                             float* __restrict__ out, int T, int Hn, int n)
{
    int i = blockIdx.x * blockDim.x + threadIdx.x;
    if (i >= n) return;
    int b = i / Hn, hh = i % Hn;
    const float* sp = s + (long long)b * T;
    const float* ep = enc + (long long)b * T * Hn + hh;
    float acc = 0.0f;
    for (int t = 0; t < T; ++t) acc += sp[t] * ep[(long long)t * Hn];
    out[i] = acc;
}

__global__ void tanh_ip(float* x, long long n) {
    long long i = (long long)blockIdx.x * blockDim.x + threadIdx.x;
    if (i < n) x[i] = tanhf(x[i]);
}

// out[m,n] = X[m,:] . W[n,:] + bias[n]   (tiny N)
__global__ void small_linear(const float* __restrict__ X, const float* __restrict__ W,
                             const float* __restrict__ bias, float* __restrict__ out,
                             int Mrows, int Nout, int Kdim)
{
    int i = blockIdx.x * blockDim.x + threadIdx.x;
    if (i >= Mrows * Nout) return;
    int m = i / Nout, n = i % Nout;
    const float* x = X + (long long)m * Kdim;
    const float* w = W + (long long)n * Kdim;
    float acc = bias[n];
    for (int k = 0; k < Kdim; ++k) acc += x[k] * w[k];
    out[i] = acc;
}

// ---------------- host orchestration ----------------
extern "C" void kernel_launch(void* const* d_in, const int* in_sizes, int n_in,
                              void* d_out, int out_size, void* d_ws, size_t ws_size,
                              hipStream_t stream)
{
    (void)in_sizes; (void)n_in; (void)out_size; (void)ws_size;

    // inputs (setup_inputs dict insertion order, params flattened in insertion order)
    const float* q_emb  = (const float*)d_in[0];
    const int*   q_len  = (const int*)d_in[1];
    const int*   q_rec  = (const int*)d_in[2];
    const float* hs_emb = (const float*)d_in[3];
    const int*   hs_len = (const int*)d_in[4];
    const int*   hs_rec = (const int*)d_in[5];
    const float* c_emb  = (const float*)d_in[6];
    const int*   c_len  = (const int*)d_in[7];
    const int*   c_rec  = (const int*)d_in[8];
    // 9..20: q_lstm, hs_lstm, col_lstm (Wih, Whh, bih, bhh)
    // 21..42: linear (w,b) pairs in insertion order

    float* out = (float*)d_out;                 // [B*4] col_num_score, then [B*C] col_score

    // workspace bump allocation (floats)
    float* ws = (float*)d_ws;
    size_t off = 0;
    auto alloc = [&](size_t nflt) { float* p = ws + off; off += (nflt + 63) & ~(size_t)63; return p; };

    float* enc_q   = alloc((size_t)BB * QQ  * HH);   // 4.19M
    float* enc_hs  = alloc((size_t)BB * HSQ * HH);   // 2.10M
    float* enc_col = alloc((size_t)BB * CC  * HH);   // 8.39M
    float* X       = alloc((size_t)BB * CC  * G4);   // 33.55M big scratch (xg, later reused)
    float* R       = alloc((size_t)BB * CC  * HH);   // 8.39M raw-enc scratch
    float* G       = alloc((size_t)BB * G4);         // hh-gates
    float* hbuf    = alloc((size_t)BB * HH);
    float* cbuf    = alloc((size_t)BB * HH);
    float* biasc   = alloc((size_t)G4);
    float* ATT     = alloc((size_t)BB * CC * QQ);    // attention scores/probs
    float* S       = alloc((size_t)BB * QQ);         // prob col-sums
    float* qw_num  = alloc((size_t)BB * HH);
    float* hsw_num = alloc((size_t)BB * HH);
    float* t_num   = alloc((size_t)BB * HH);

    // sub-regions of X reused after the LSTM phase
    const size_t BCH = (size_t)BB * CC * HH;         // 8,388,608
    float* QW  = X;                                  // [B*C, H]
    float* HSW = X + BCH;                            // [B*C, H]
    float* TT  = X + 2 * BCH;                        // [B*C, H]
    float* LIN = X + 3 * BCH;                        // transient linear outputs (<=4.19M)

    auto gemm = [&](const float* A, const float* Bmat, const float* bias, const float* bias2,
                    float* Cp, int M, int N, int K, int lda, int ldb, int ldc,
                    long long sA, long long sB, long long sC, int nbatch,
                    int b_is_nk, int accum) {
        dim3 grid((N + BN - 1) / BN, (M + BM - 1) / BM, nbatch);
        wmma_gemm<<<grid, dim3(128), 0, stream>>>(A, Bmat, bias, bias2, Cp,
                                                  M, N, K, lda, ldb, ldc,
                                                  sA, sB, sC, b_is_nk, accum);
    };
    auto lin1d = [&](int n) { return dim3((n + 255) / 256); };

    // ---------------- Phase 1: three LSTMs ----------------
    struct Cfg { const float* x; int T; int wi, wh, b1, b2; const int* rec; float* enc; };
    Cfg cfgs[3] = {
        { q_emb,  QQ,  9, 10, 11, 12, q_rec,  enc_q   },
        { hs_emb, HSQ, 13, 14, 15, 16, hs_rec, enc_hs  },
        { c_emb,  CC,  17, 18, 19, 20, c_rec,  enc_col },
    };

    for (int li = 0; li < 3; ++li) {
        const Cfg& cg = cfgs[li];
        const float* Wih = (const float*)d_in[cg.wi];   // [4H, D]
        const float* Whh = (const float*)d_in[cg.wh];   // [4H, H]
        const float* bih = (const float*)d_in[cg.b1];
        const float* bhh = (const float*)d_in[cg.b2];

        add2_f32<<<lin1d(G4), 256, 0, stream>>>(biasc, bih, bhh, G4);

        // xg = x @ Wih^T + (bih+bhh)   : [B*T, 4H]
        gemm(cg.x, Wih, biasc, nullptr, X,
             BB * cg.T, G4, DD, DD, DD, G4, 0, 0, 0, 1, /*nk*/1, /*acc*/0);

        zero_f32<<<lin1d(BB * HH), 256, 0, stream>>>(hbuf, BB * HH);
        zero_f32<<<lin1d(BB * HH), 256, 0, stream>>>(cbuf, BB * HH);

        for (int t = 0; t < cg.T; ++t) {
            // G = h @ Whh^T : [B, 4H]
            gemm(hbuf, Whh, nullptr, nullptr, G,
                 BB, G4, HH, HH, HH, G4, 0, 0, 0, 1, 1, 0);
            lstm_cell<<<lin1d(BB * HH), 256, 0, stream>>>(
                X + (long long)t * G4, (long long)cg.T * G4,
                G, cbuf, hbuf,
                R + (long long)t * HH, (long long)cg.T * HH,
                BB, HH);
        }
        long long n = (long long)BB * cg.T * HH;
        gather_rows<<<lin1d((int)n), 256, 0, stream>>>(cg.enc, R, cg.rec,
                                                       (long long)cg.T * HH, n);
    }

    // ---------------- Phase 2: num-of-columns head ----------------
    // lin_q = q_enc @ q_num_att.w^T + b
    gemm(enc_q, (const float*)d_in[21], (const float*)d_in[22], nullptr, LIN,
         BB * QQ, HH, HH, HH, HH, HH, 0, 0, 0, 1, 1, 0);
    // att_qc_num[b] = col_enc[b] @ lin_q[b]^T : [C, Q]
    gemm(enc_col, LIN, nullptr, nullptr, ATT,
         CC, QQ, HH, HH, HH, QQ,
         (long long)CC * HH, (long long)QQ * HH, (long long)CC * QQ, BB, 1, 0);
    softmax_mask<<<BB * CC, 64, 0, stream>>>(ATT, CC, QQ, c_len, c_rec, q_len, q_rec);
    sum_over_c<<<lin1d(BB * QQ), 256, 0, stream>>>(ATT, S, CC, QQ, BB * QQ);
    weighted_sum<<<lin1d(BB * HH), 256, 0, stream>>>(S, enc_q, qw_num, QQ, HH, BB * HH);

    // hs side
    gemm(enc_hs, (const float*)d_in[23], (const float*)d_in[24], nullptr, LIN,
         BB * HSQ, HH, HH, HH, HH, HH, 0, 0, 0, 1, 1, 0);
    gemm(enc_col, LIN, nullptr, nullptr, ATT,
         CC, HSQ, HH, HH, HH, HSQ,
         (long long)CC * HH, (long long)HSQ * HH, (long long)CC * HSQ, BB, 1, 0);
    softmax_mask<<<BB * CC, 64, 0, stream>>>(ATT, CC, HSQ, c_len, c_rec, hs_len, hs_rec);
    sum_over_c<<<lin1d(BB * HSQ), 256, 0, stream>>>(ATT, S, CC, HSQ, BB * HSQ);
    weighted_sum<<<lin1d(BB * HH), 256, 0, stream>>>(S, enc_hs, hsw_num, HSQ, HH, BB * HH);

    // t_num = tanh(qw_num @ Wq^T + bq + hsw_num @ Whs^T + bhs)
    gemm(qw_num, (const float*)d_in[25], (const float*)d_in[26], nullptr, t_num,
         BB, HH, HH, HH, HH, HH, 0, 0, 0, 1, 1, 0);
    gemm(hsw_num, (const float*)d_in[27], (const float*)d_in[28], nullptr, t_num,
         BB, HH, HH, HH, HH, HH, 0, 0, 0, 1, 1, 1);
    tanh_ip<<<lin1d(BB * HH), 256, 0, stream>>>(t_num, (long long)BB * HH);
    // col_num_score = t_num @ W^T + b : [B, 4] -> out[0:512]
    small_linear<<<lin1d(BB * NLAB), 256, 0, stream>>>(
        t_num, (const float*)d_in[29], (const float*)d_in[30], out, BB, NLAB, HH);

    // ---------------- Phase 3: per-column head ----------------
    // q attention
    gemm(enc_q, (const float*)d_in[31], (const float*)d_in[32], nullptr, LIN,
         BB * QQ, HH, HH, HH, HH, HH, 0, 0, 0, 1, 1, 0);
    gemm(enc_col, LIN, nullptr, nullptr, ATT,
         CC, QQ, HH, HH, HH, QQ,
         (long long)CC * HH, (long long)QQ * HH, (long long)CC * QQ, BB, 1, 0);
    softmax_mask<<<BB * CC, 64, 0, stream>>>(ATT, CC, QQ, c_len, c_rec, q_len, q_rec);
    // q_weighted[b] = prob[b] @ q_enc[b] : [C, H]  (NN gemm)
    gemm(ATT, enc_q, nullptr, nullptr, QW,
         CC, HH, QQ, QQ, HH, HH,
         (long long)CC * QQ, (long long)QQ * HH, (long long)CC * HH, BB, 0, 0);

    // hs attention
    gemm(enc_hs, (const float*)d_in[33], (const float*)d_in[34], nullptr, LIN,
         BB * HSQ, HH, HH, HH, HH, HH, 0, 0, 0, 1, 1, 0);
    gemm(enc_col, LIN, nullptr, nullptr, ATT,
         CC, HSQ, HH, HH, HH, HSQ,
         (long long)CC * HH, (long long)HSQ * HH, (long long)CC * HSQ, BB, 1, 0);
    softmax_mask<<<BB * CC, 64, 0, stream>>>(ATT, CC, HSQ, c_len, c_rec, hs_len, hs_rec);
    gemm(ATT, enc_hs, nullptr, nullptr, HSW,
         CC, HH, HSQ, HSQ, HH, HH,
         (long long)CC * HSQ, (long long)HSQ * HH, (long long)CC * HH, BB, 0, 0);

    // TT = tanh(QW@Wq^T+bq + HSW@Whs^T+bhs + col_enc@Wc^T+bc)
    gemm(QW, (const float*)d_in[35], (const float*)d_in[36], nullptr, TT,
         BB * CC, HH, HH, HH, HH, HH, 0, 0, 0, 1, 1, 0);
    gemm(HSW, (const float*)d_in[39], (const float*)d_in[40], nullptr, TT,
         BB * CC, HH, HH, HH, HH, HH, 0, 0, 0, 1, 1, 1);
    gemm(enc_col, (const float*)d_in[37], (const float*)d_in[38], nullptr, TT,
         BB * CC, HH, HH, HH, HH, HH, 0, 0, 0, 1, 1, 1);
    tanh_ip<<<lin1d(BB * CC * 2), 512, 0, stream>>>(TT, (long long)BB * CC * HH);

    // col_score = TT @ w^T + b : [B*C] -> out[512:]
    small_linear<<<lin1d(BB * CC), 256, 0, stream>>>(
        TT, (const float*)d_in[41], (const float*)d_in[42], out + BB * NLAB,
        BB * CC, 1, HH);
}
